// CenterAttention_14791867367785
// MI455X (gfx1250) — matrix operand order repaired and checked by reading, not compiled, not run on hardware
//
#include <hip/hip_runtime.h>
#include <hip/hip_bf16.h>

typedef __attribute__((ext_vector_type(16))) _Float16 v16h;
typedef __attribute__((ext_vector_type(8)))  _Float16 v8h;
typedef __attribute__((ext_vector_type(8)))  float    v8f;

union HU { v16h v; v8h h[2]; };

// A-fragment (16x32 f16): lane m=lane&15, kb=(lane&16)?8:0.
// halves 0..7 = K kb..kb+7 at p, halves 8..15 = K kb+16..kb+23 at p+16.
__device__ __forceinline__ v16h ldA(const _Float16* p) {
    HU u;
    u.h[0] = *(const v8h*)p;
    u.h[1] = *(const v8h*)(p + 16);
    return u.v;
}
// B-fragment (32x16 f16): lane n=lane&15, kb=(lane&16)?8:0.
// halves 0..7 = K kb..kb+7 at p, halves 8..15 = K kb+16..kb+23 at p+16.
__device__ __forceinline__ v16h ldB(const _Float16* p) {
    HU u;
    u.h[0] = *(const v8h*)p;
    u.h[1] = *(const v8h*)(p + 16);
    return u.v;
}

// ---- packed weight layout in d_ws (halves unless noted) ----
// W1p: [112][72]  @ 0       (8064 halves)
// W2p: [112][136] @ 8064    (15232 halves)
// W3p: [64][136]  @ 23296   (8704 halves)   -> total 32000 halves = 64000 B
// then (as float, i.e. wsF + 16000): s1[112], b1[112], s2[112], b2[112]
#define W2_OFF 8064
#define W3_OFF 23296
#define SB_OFF 16000   // float index from base

__global__ void ca_pack(const float* __restrict__ W1,
                        const float* __restrict__ g1, const float* __restrict__ be1,
                        const float* __restrict__ rm1, const float* __restrict__ rv1,
                        const float* __restrict__ W2,
                        const float* __restrict__ g2, const float* __restrict__ be2,
                        const float* __restrict__ rm2, const float* __restrict__ rv2,
                        const float* __restrict__ W3,
                        _Float16* __restrict__ wsH, float* __restrict__ wsF)
{
    int idx = blockIdx.x * blockDim.x + threadIdx.x;
    int stride = gridDim.x * blockDim.x;
    for (int t = idx; t < 32448; t += stride) {
        if (t < W2_OFF) {
            int o = t / 72, k = t - o * 72;
            float v = (o < 98 && k < 49) ? W1[o * 49 + k] : 0.f;
            wsH[t] = (_Float16)v;
        } else if (t < W3_OFF) {
            int r = t - W2_OFF; int o = r / 136, k = r - o * 136;
            float v = (o < 98 && k < 98) ? W2[o * 98 + k] : 0.f;
            wsH[t] = (_Float16)v;
        } else if (t < 32000) {
            int r = t - W3_OFF; int o = r / 136, k = r - o * 136;
            float v = (o < 49 && k < 98) ? W3[o * 98 + k] : 0.f;
            wsH[t] = (_Float16)v;
        } else {
            int r = t - 32000; int sel = r / 112, ch = r - sel * 112;
            float s = 0.f, bo = 0.f;
            if (ch < 98) {
                if (sel < 2) { s = g1[ch] * rsqrtf(rv1[ch] + 1e-5f); bo = be1[ch] - rm1[ch] * s; }
                else         { s = g2[ch] * rsqrtf(rv2[ch] + 1e-5f); bo = be2[ch] - rm2[ch] * s; }
            }
            float* dst = wsF + SB_OFF;
            if      (sel == 0) dst[ch]        = s;
            else if (sel == 1) dst[112 + ch]  = bo;
            else if (sel == 2) dst[224 + ch]  = s;
            else               dst[336 + ch]  = bo;
        }
    }
}

// block = 256 threads (8 waves); handles one (b, d, 128-pixel strip).
// wave w owns the 16-pixel N-tile [w*16, w*16+16).
#define BSTR 136  // LDS activation row stride in halves (272B: 16B-aligned, conflict-free)

__global__ void __launch_bounds__(256)
ca_main(const float* __restrict__ x,
        const _Float16* __restrict__ wsH,
        const float* __restrict__ wsF,
        float* __restrict__ out)
{
    extern __shared__ char smem[];
    float*    xs   = (float*)smem;                         // 1024 f32 = 4096 B
    _Float16* buf0 = (_Float16*)(smem + 4096);             // [128][136] f16
    _Float16* buf1 = buf0 + 128 * BSTR;                    // [128][136] f16

    const int tid   = threadIdx.x;
    const int bid   = blockIdx.x;
    const int strip = bid & 7;
    const int d     = (bid >> 3) % 9;
    const int b     = bid / 72;

    // stage x(b) into LDS
    const float* xb = x + (size_t)b * 1024;
    for (int i = tid; i < 1024; i += 256) xs[i] = xb[i];
    __syncthreads();

    // build T (49-vector per pixel, padded K to 64) and zero pad columns of both buffers
    const int dl = (d < 4) ? (4 - d) : (d - 4);
    for (int idx = tid; idx < 128 * 64; idx += 256) {
        int pix = idx >> 6, k = idx & 63;
        float v = 0.f;
        if (k < 49 && d != 4) {
            int i7 = k / 7, j7 = k - i7 * 7;
            int dh = (i7 - 3) * dl, dw = (j7 - 3) * dl;
            int gp = strip * 128 + pix;
            int h = (gp >> 5) + dh, w = (gp & 31) + dw;
            float aV = ((unsigned)h < 32u && (unsigned)w < 32u) ? xs[h * 32 + w] : 0.f;
            int hc = 16 + dh, wc = dw;
            float cV = ((unsigned)hc < 32u && (unsigned)wc < 32u) ? xs[hc * 32 + wc] : 0.f;
            v = aV - cV;
        }
        buf0[pix * BSTR + k] = (_Float16)v;
    }
    for (int idx = tid; idx < 128 * 64; idx += 256) {
        int pix = idx >> 6, k = 64 + (idx & 63);
        buf0[pix * BSTR + k] = (_Float16)0.f;
        buf1[pix * BSTR + k] = (_Float16)0.f;
    }
    __syncthreads();

    const int lane = tid & 31;
    const int wv   = tid >> 5;
    const int n    = lane & 15;                 // A row m / B column n / C column n
    const int roff = (lane & 16) ? 8 : 0;       // A kb, and C row offset
    const int kbB  = (lane & 16) ? 8 : 0;       // B kb

    const _Float16* Brow0 = buf0 + (wv * 16 + n) * BSTR;
    const _Float16* Brow1 = buf1 + (wv * 16 + n) * BSTR;
    const float* s1 = wsF + SB_OFF;
    const float* bb1 = s1 + 112;
    const float* s2 = bb1 + 112;
    const float* bb2 = s2 + 112;

    // ---- layer 1: 112x128x64, T(buf0) -> buf1 ----
    #pragma unroll
    for (int mt = 0; mt < 7; ++mt) {
        v8f acc = {0.f,0.f,0.f,0.f,0.f,0.f,0.f,0.f};
        #pragma unroll
        for (int ks = 0; ks < 2; ++ks) {
            v16h a  = ldA(wsH + (mt * 16 + n) * 72 + ks * 32 + roff);
            v16h bf = ldB(Brow0 + ks * 32 + kbB);
            acc = __builtin_amdgcn_wmma_f32_16x16x32_f16(false, a, false, bf,
                                                         (short)0, acc, false, false);
        }
        int ch0 = mt * 16 + roff;
        float4 sA = *(const float4*)(s1 + ch0);
        float4 sB = *(const float4*)(s1 + ch0 + 4);
        float4 bA = *(const float4*)(bb1 + ch0);
        float4 bB = *(const float4*)(bb1 + ch0 + 4);
        float sv[8] = {sA.x,sA.y,sA.z,sA.w,sB.x,sB.y,sB.z,sB.w};
        float bv[8] = {bA.x,bA.y,bA.z,bA.w,bB.x,bB.y,bB.z,bB.w};
        #pragma unroll
        for (int r = 0; r < 8; ++r) {
            int ch = ch0 + r;
            if (ch < 98) {
                float y = acc[r] * sv[r] + bv[r];
                y = (y > 0.f) ? y : 0.01f * y;
                buf1[(wv * 16 + n) * BSTR + ch] = (_Float16)y;
            }
        }
    }
    __syncthreads();

    // ---- layer 2: 112x128x128, buf1 -> buf0 ----
    #pragma unroll
    for (int mt = 0; mt < 7; ++mt) {
        v8f acc = {0.f,0.f,0.f,0.f,0.f,0.f,0.f,0.f};
        #pragma unroll
        for (int ks = 0; ks < 4; ++ks) {
            v16h a  = ldA(wsH + W2_OFF + (mt * 16 + n) * 136 + ks * 32 + roff);
            v16h bf = ldB(Brow1 + ks * 32 + kbB);
            acc = __builtin_amdgcn_wmma_f32_16x16x32_f16(false, a, false, bf,
                                                         (short)0, acc, false, false);
        }
        int ch0 = mt * 16 + roff;
        float4 sA = *(const float4*)(s2 + ch0);
        float4 sB = *(const float4*)(s2 + ch0 + 4);
        float4 bA = *(const float4*)(bb2 + ch0);
        float4 bB = *(const float4*)(bb2 + ch0 + 4);
        float sv[8] = {sA.x,sA.y,sA.z,sA.w,sB.x,sB.y,sB.z,sB.w};
        float bv[8] = {bA.x,bA.y,bA.z,bA.w,bB.x,bB.y,bB.z,bB.w};
        #pragma unroll
        for (int r = 0; r < 8; ++r) {
            int ch = ch0 + r;
            if (ch < 98) {
                float y = acc[r] * sv[r] + bv[r];
                y = (y > 0.f) ? y : 0.01f * y;
                buf0[(wv * 16 + n) * BSTR + ch] = (_Float16)y;
            }
        }
    }
    __syncthreads();

    // ---- layer 3: 64x128x128, buf0 -> sigmoid -> global ----
    const int pixG = strip * 128 + wv * 16 + n;
    #pragma unroll
    for (int mt = 0; mt < 4; ++mt) {
        v8f acc = {0.f,0.f,0.f,0.f,0.f,0.f,0.f,0.f};
        #pragma unroll
        for (int ks = 0; ks < 4; ++ks) {
            v16h a  = ldA(wsH + W3_OFF + (mt * 16 + n) * 136 + ks * 32 + roff);
            v16h bf = ldB(Brow0 + ks * 32 + kbB);
            acc = __builtin_amdgcn_wmma_f32_16x16x32_f16(false, a, false, bf,
                                                         (short)0, acc, false, false);
        }
        int ch0 = mt * 16 + roff;
        #pragma unroll
        for (int r = 0; r < 8; ++r) {
            int ch = ch0 + r;
            if (ch < 49) {
                float y = acc[r];
                y = 1.f / (1.f + expf(-y));
                out[(((size_t)b * 49 + ch) * 9 + d) * 1024 + pixG] = y;
            }
        }
    }
}

extern "C" void kernel_launch(void* const* d_in, const int* in_sizes, int n_in,
                              void* d_out, int out_size, void* d_ws, size_t ws_size,
                              hipStream_t stream) {
    const float* x   = (const float*)d_in[0];
    const float* W1  = (const float*)d_in[1];
    const float* g1  = (const float*)d_in[2];
    const float* b1  = (const float*)d_in[3];
    const float* rm1 = (const float*)d_in[4];
    const float* rv1 = (const float*)d_in[5];
    const float* W2  = (const float*)d_in[6];
    const float* g2  = (const float*)d_in[7];
    const float* b2  = (const float*)d_in[8];
    const float* rm2 = (const float*)d_in[9];
    const float* rv2 = (const float*)d_in[10];
    const float* W3  = (const float*)d_in[11];

    _Float16* wsH = (_Float16*)d_ws;   // needs ~66 KB of d_ws
    float*    wsF = (float*)d_ws;

    ca_pack<<<64, 256, 0, stream>>>(W1, g1, b1, rm1, rv1,
                                    W2, g2, b2, rm2, rv2, W3, wsH, wsF);

    // 256 batches * 9 dil-channels * 8 strips = 18432 blocks
    const int smem = 4096 + 2 * 128 * BSTR * 2;  // 73728 B
    ca_main<<<18432, 256, smem, stream>>>(x, wsH, wsF, (float*)d_out);
}
